// AttentionEncoderToFixedLength_32693291057693
// MI455X (gfx1250) — compile-verified, matrix operand order checked by reading, not run on hardware
//
#include <hip/hip_runtime.h>
#include <hip/hip_bf16.h>

// ---------------------------------------------------------------------------
// MI455X (gfx1250) implementation: every matmul runs on V_WMMA_F32_16X16X32_BF16
// (wave32, fp32 accumulate). ~210 GFLOP total vs ~15us of HBM traffic at
// 23.3 TB/s -> compute bound -> all dense math on the WMMA pipe.
// Fast GEMM path is guard-free, 32x64 per wave (8 WMMAs / 6 fragment loads),
// and explicitly double-buffered so WMMAs consume fragments loaded one full
// K-iteration earlier (no s_wait_loadcnt 0x0 on the critical path).
// ---------------------------------------------------------------------------

typedef __attribute__((ext_vector_type(16))) __bf16 v16bf;
typedef __attribute__((ext_vector_type(8)))  __bf16 v8bf;
typedef __attribute__((ext_vector_type(8)))  float  v8f;

#define DEVINL __device__ __forceinline__

// Load a 16x32 bf16 WMMA A/B fragment from a row-major [row, K] matrix.
// ISA layout (16-bit A 16x32): lanes 0-15 hold K = k0+{0..7, 16..23},
// lanes 16-31 hold K = k0+{8..15, 24..31}; row = lane & 15.
DEVINL v16bf load_frag_nc(const __bf16* base, int ld, int row, int k0) {
  const int lane = threadIdx.x & 31;
  const int half = lane >> 4;
  const __bf16* p = base + (size_t)row * ld + k0 + half * 8;
  v8bf lo = *(const v8bf*)(p);
  v8bf hi = *(const v8bf*)(p + 16);
  v16bf f;
#pragma unroll
  for (int i = 0; i < 8; ++i) { f[i] = lo[i]; f[i + 8] = hi[i]; }
  return f;
}

DEVINL v16bf load_frag_guard(const __bf16* base, int ld, int row, int k0, bool valid) {
  v16bf f;
  if (valid) {
    f = load_frag_nc(base, ld, row, k0);
  } else {
#pragma unroll
    for (int i = 0; i < 16; ++i) f[i] = (__bf16)0.0f;
  }
  return f;
}

DEVINL v8f wmma_bf16(v16bf a, v16bf b, v8f c) {
  return __builtin_amdgcn_wmma_f32_16x16x32_bf16(
      /*neg_a=*/false, a, /*neg_b=*/false, b,
      /*c_mod=*/(short)0, c, /*reuse_a=*/false, /*reuse_b=*/false);
}

DEVINL float apply_act(float v, int act) {
  if (act == 1)      return tanhf(v);
  else if (act == 2) return fmaxf(v, 0.0f);
  return v;
}

// ---------------------------------------------------------------------------
// Fast WMMA GEMM (no guards): C[R,N] = act(A[R,K] * W[N,K]^T + bias).
// Requires R % 256 == 0, N % 64 == 0, K % 32 == 0.
// 8 waves; each wave owns a 32x64 strip. Double-buffered K loop: iteration k
// issues all 12 b128 fragment loads for k+32, then runs 8 WMMAs on the
// fragments loaded during iteration k-32. global_prefetch_b8 runs 2 slices
// ahead of the double buffer.
// ---------------------------------------------------------------------------
__global__ __launch_bounds__(256)
void gemm_full_wmma(const __bf16* __restrict__ A, const __bf16* __restrict__ W,
                    const float* __restrict__ Cbias,
                    float* __restrict__ Cf, __bf16* __restrict__ Cb,
                    int R, int N, int K, int act) {
  const int lane = threadIdx.x & 31;
  const int half = lane >> 4;
  const int col  = lane & 15;
  const int wave = threadIdx.x >> 5;
  const int r0 = blockIdx.x * 256 + wave * 32;
  const int c0 = blockIdx.y * 64;

  v8f acc0[4], acc1[4];
#pragma unroll
  for (int j = 0; j < 4; ++j)
#pragma unroll
    for (int g = 0; g < 8; ++g) { acc0[j][g] = 0.0f; acc1[j][g] = 0.0f; }

  const int ar0 = r0 + col;
  const int ar1 = r0 + 16 + col;

  // Prologue: load K-slice 0.
  v16bf a0 = load_frag_nc(A, K, ar0, 0);
  v16bf a1 = load_frag_nc(A, K, ar1, 0);
  v16bf b0 = load_frag_nc(W, K, c0 + col, 0);
  v16bf b1 = load_frag_nc(W, K, c0 + 16 + col, 0);
  v16bf b2 = load_frag_nc(W, K, c0 + 32 + col, 0);
  v16bf b3 = load_frag_nc(W, K, c0 + 48 + col, 0);

  for (int k0 = 0; k0 < K; k0 += 32) {
    // Next K-slice (clamped to 0 on the last iteration: harmless redundant
    // load, keeps the loop branch-free and the buffers always defined).
    const int kn = (k0 + 32 < K) ? (k0 + 32) : 0;
    if (k0 + 64 < K) {  // uniform branch: prefetch 2 slices ahead
      __builtin_prefetch((const void*)(A + (size_t)ar0 * K + k0 + 64), 0, 1);
      __builtin_prefetch((const void*)(W + (size_t)(c0 + col) * K + k0 + 64), 0, 1);
    }
    v16bf na0 = load_frag_nc(A, K, ar0, kn);
    v16bf na1 = load_frag_nc(A, K, ar1, kn);
    v16bf nb0 = load_frag_nc(W, K, c0 + col, kn);
    v16bf nb1 = load_frag_nc(W, K, c0 + 16 + col, kn);
    v16bf nb2 = load_frag_nc(W, K, c0 + 32 + col, kn);
    v16bf nb3 = load_frag_nc(W, K, c0 + 48 + col, kn);

    acc0[0] = wmma_bf16(a0, b0, acc0[0]);
    acc1[0] = wmma_bf16(a1, b0, acc1[0]);
    acc0[1] = wmma_bf16(a0, b1, acc0[1]);
    acc1[1] = wmma_bf16(a1, b1, acc1[1]);
    acc0[2] = wmma_bf16(a0, b2, acc0[2]);
    acc1[2] = wmma_bf16(a1, b2, acc1[2]);
    acc0[3] = wmma_bf16(a0, b3, acc0[3]);
    acc1[3] = wmma_bf16(a1, b3, acc1[3]);

    a0 = na0; a1 = na1; b0 = nb0; b1 = nb1; b2 = nb2; b3 = nb3;
  }

#pragma unroll
  for (int j = 0; j < 4; ++j) {
    const int n = c0 + j * 16 + col;
    const float bv = Cbias ? Cbias[n] : 0.0f;
#pragma unroll
    for (int g = 0; g < 8; ++g) {
      const int ra = r0 + g + half * 8;
      const int rb = ra + 16;
      const float va = apply_act(acc0[j][g] + bv, act);
      const float vb = apply_act(acc1[j][g] + bv, act);
      if (Cf) {
        Cf[(size_t)ra * N + n] = va;
        Cf[(size_t)rb * N + n] = vb;
      }
      if (Cb) {
        Cb[(size_t)ra * N + n] = (__bf16)va;
        Cb[(size_t)rb * N + n] = (__bf16)vb;
      }
    }
  }
}

// ---------------------------------------------------------------------------
// Edge WMMA GEMM (row/col guarded) for small R or N (logits N=8, head R=8).
// 8 waves; each wave: 16 rows x 64 cols.
// ---------------------------------------------------------------------------
__global__ __launch_bounds__(256)
void gemm_edge_wmma(const __bf16* __restrict__ A, const __bf16* __restrict__ W,
                    const float* __restrict__ Cbias,
                    float* __restrict__ Cf, __bf16* __restrict__ Cb,
                    int R, int N, int K, int act) {
  const int lane = threadIdx.x & 31;
  const int half = lane >> 4;
  const int col  = lane & 15;
  const int wave = threadIdx.x >> 5;
  const int r0 = blockIdx.x * 128 + wave * 16;
  const int c0 = blockIdx.y * 64;

  v8f acc[4];
#pragma unroll
  for (int j = 0; j < 4; ++j)
#pragma unroll
    for (int g = 0; g < 8; ++g) acc[j][g] = 0.0f;

  const int arow = r0 + col;
  const bool avalid = arow < R;

  for (int k0 = 0; k0 < K; k0 += 32) {
    v16bf af = load_frag_guard(A, K, arow, k0, avalid);
#pragma unroll
    for (int j = 0; j < 4; ++j) {
      const int n = c0 + j * 16 + col;
      v16bf bf = load_frag_guard(W, K, n, k0, n < N);
      acc[j] = wmma_bf16(af, bf, acc[j]);
    }
  }

#pragma unroll
  for (int j = 0; j < 4; ++j) {
    const int n = c0 + j * 16 + col;
    if (n >= N) continue;
    const float bv = Cbias ? Cbias[n] : 0.0f;
#pragma unroll
    for (int g = 0; g < 8; ++g) {
      const int r = r0 + g + half * 8;
      if (r >= R) continue;
      const float v = apply_act(acc[j][g] + bv, act);
      if (Cf) Cf[(size_t)r * N + n] = v;
      if (Cb) Cb[(size_t)r * N + n] = (__bf16)v;
    }
  }
}

// ---------------------------------------------------------------------------
// Flash-attention with WMMA. Q,K in [B,T,H,E] bf16; V pre-transposed to
// [B,H,E,T] bf16 so its B-fragments load contiguously. Output att in
// [B,T,H*E] bf16 (= reference transpose(0,2,1,3).reshape). 4 waves/block,
// each wave owns 16 query rows; online softmax per 32-key block; P goes
// through LDS to re-shape into A-fragment layout. All fragment loads are
// guard-free: T % 32 == 0 guarantees srow < T for every processed block;
// key masking is applied to the fp32 scores instead.
// ---------------------------------------------------------------------------
__global__ __launch_bounds__(128)
void attn_flash_wmma(const __bf16* __restrict__ Qb, const __bf16* __restrict__ Kb,
                     const __bf16* __restrict__ Vt, const int* __restrict__ lens,
                     __bf16* __restrict__ attb, int Bn, int Hn, int Tn) {
  const int En = 64, HE = 512;
  const int lane = threadIdx.x & 31;
  const int half = lane >> 4;
  const int col  = lane & 15;
  const int wv   = threadIdx.x >> 5;

  const int bh = blockIdx.y;
  const int b = bh / Hn, h = bh % Hn;
  const int t0 = blockIdx.x * 64 + wv * 16;
  const int len = lens[b];

  const __bf16* Qbase = Qb + (size_t)b * Tn * HE + h * En;   // row t, ld HE
  const __bf16* Kbase = Kb + (size_t)b * Tn * HE + h * En;   // row s, ld HE
  const __bf16* Vbase = Vt + ((size_t)b * Hn + h) * En * Tn; // row e, ld Tn

  // Q fragments stay resident (E=64 -> two K=32 fragments).
  v16bf qf0 = load_frag_nc(Qbase, HE, t0 + col, 0);
  v16bf qf1 = load_frag_nc(Qbase, HE, t0 + col, 32);

  float m[8], l[8];
  v8f o[4];
#pragma unroll
  for (int g = 0; g < 8; ++g) { m[g] = -3.0e38f; l[g] = 0.0f; }
#pragma unroll
  for (int j = 0; j < 4; ++j)
#pragma unroll
    for (int g = 0; g < 8; ++g) o[j][g] = 0.0f;

  __shared__ __align__(16) __bf16 Pl[4][16][32];

  for (int s0 = 0; s0 < len; s0 += 32) {
    float sv[2][8];
#pragma unroll
    for (int sb = 0; sb < 2; ++sb) {
      const int srow = s0 + sb * 16 + col;
      v16bf kf0 = load_frag_nc(Kbase, HE, srow, 0);
      v16bf kf1 = load_frag_nc(Kbase, HE, srow, 32);
      v8f s;
#pragma unroll
      for (int g = 0; g < 8; ++g) s[g] = 0.0f;
      s = wmma_bf16(qf0, kf0, s);
      s = wmma_bf16(qf1, kf1, s);
#pragma unroll
      for (int g = 0; g < 8; ++g) {
        float x = s[g] * 0.125f;           // 1/sqrt(E)
        if (srow >= len) x = -3.0e38f;     // key mask
        sv[sb][g] = x;
      }
    }

    float p0[8], p1[8];
#pragma unroll
    for (int g = 0; g < 8; ++g) {
      float mx = fmaxf(sv[0][g], sv[1][g]);
#pragma unroll
      for (int off = 8; off; off >>= 1) mx = fmaxf(mx, __shfl_xor(mx, off, 16));
      const float mnew = fmaxf(m[g], mx);
      const float alpha = __expf(m[g] - mnew);
      m[g] = mnew;
      p0[g] = __expf(sv[0][g] - mnew);
      p1[g] = __expf(sv[1][g] - mnew);
      float sum = p0[g] + p1[g];
#pragma unroll
      for (int off = 8; off; off >>= 1) sum += __shfl_xor(sum, off, 16);
      l[g] = l[g] * alpha + sum;
#pragma unroll
      for (int j = 0; j < 4; ++j) o[j][g] *= alpha;
    }

    // Stage P into LDS row-major [16 q][32 s] so it can be re-read in
    // A-fragment layout for the P*V WMMA.
#pragma unroll
    for (int g = 0; g < 8; ++g) {
      const int row = g + half * 8;
      Pl[wv][row][col]      = (__bf16)p0[g];
      Pl[wv][row][16 + col] = (__bf16)p1[g];
    }
    __syncthreads();
    v16bf pf = load_frag_nc((const __bf16*)&Pl[wv][0][0], 32, col, 0);
#pragma unroll
    for (int j = 0; j < 4; ++j) {
      const int e = j * 16 + col;
      v16bf vf = load_frag_nc(Vbase, Tn, e, s0);  // [E,T]: contiguous in s
      o[j] = wmma_bf16(pf, vf, o[j]);
    }
    __syncthreads();
  }

#pragma unroll
  for (int j = 0; j < 4; ++j) {
#pragma unroll
    for (int g = 0; g < 8; ++g) {
      const int t = t0 + g + half * 8;
      const int e = j * 16 + col;
      const float val = o[j][g] / l[g];
      attb[((size_t)b * Tn + t) * HE + h * En + e] = (__bf16)val;
    }
  }
}

// ---------------------------------------------------------------------------
// Elementwise / reduction helpers
// ---------------------------------------------------------------------------
__global__ void cvt_bf16(const float* __restrict__ src, __bf16* __restrict__ dst,
                         size_t n) {
  size_t i = (size_t)blockIdx.x * 256 + threadIdx.x;
  if (i < n) dst[i] = (__bf16)src[i];
}

// dst[r, kd] = (k < ks) ? src[r, ks] : 0   (K padding + bf16 convert)
__global__ void pad_cvt_bf16(const float* __restrict__ src, __bf16* __restrict__ dst,
                             int rows, int ks, int kd) {
  size_t i = (size_t)blockIdx.x * 256 + threadIdx.x;
  if (i >= (size_t)rows * kd) return;
  int k = (int)(i % kd);
  size_t r = i / kd;
  dst[i] = (k < ks) ? (__bf16)src[r * ks + k] : (__bf16)0.0f;
}

// h = to_hidden_out + pos_encoding(t,d), writes fp32 + bf16.
__global__ void peadd_kernel(const float* __restrict__ toh, float* __restrict__ hf,
                             __bf16* __restrict__ hb, int Tn, size_t n) {
  size_t i = (size_t)blockIdx.x * 256 + threadIdx.x;
  if (i >= n) return;
  int d = (int)(i % 512);
  int t = (int)((i / 512) % Tn);
  int k = d >> 1;
  // inv = 10000^(k/256); ang = t / inv
  float ang = (float)t * __expf(-(float)k * (9.21034037197618f / 256.0f));
  float pe = (d & 1) ? __cosf(ang) : __sinf(ang);
  float v = toh[i] + pe;
  hf[i] = v;
  hb[i] = (__bf16)v;
}

// out = LayerNorm(x + y) * g + b  (D = 512), one row per 256-thread block.
__global__ __launch_bounds__(256)
void ln_kernel(const float* __restrict__ x, const float* __restrict__ y,
               const float* __restrict__ gw, const float* __restrict__ bw,
               float* __restrict__ outf, __bf16* __restrict__ outb) {
  const int r = blockIdx.x, tid = threadIdx.x;
  const float* xr = x + (size_t)r * 512;
  float v0 = xr[tid], v1 = xr[tid + 256];
  if (y) {
    const float* yr = y + (size_t)r * 512;
    v0 += yr[tid]; v1 += yr[tid + 256];
  }
  __shared__ float red[256];
  red[tid] = v0 + v1;
  __syncthreads();
  for (int s = 128; s; s >>= 1) { if (tid < s) red[tid] += red[tid + s]; __syncthreads(); }
  const float mean = red[0] * (1.0f / 512.0f);
  __syncthreads();
  const float d0 = v0 - mean, d1 = v1 - mean;
  red[tid] = d0 * d0 + d1 * d1;
  __syncthreads();
  for (int s = 128; s; s >>= 1) { if (tid < s) red[tid] += red[tid + s]; __syncthreads(); }
  const float inv = rsqrtf(red[0] * (1.0f / 512.0f) + 1e-5f);
  const float o0 = d0 * inv * gw[tid] + bw[tid];
  const float o1 = d1 * inv * gw[tid + 256] + bw[tid + 256];
  if (outf) { float* p = outf + (size_t)r * 512; p[tid] = o0; p[tid + 256] = o1; }
  if (outb) { __bf16* p = outb + (size_t)r * 512; p[tid] = (__bf16)o0; p[tid + 256] = (__bf16)o1; }
}

// V [B,T,H,E] -> Vt [B,H,E,T]
__global__ void transpose_v(const __bf16* __restrict__ v, __bf16* __restrict__ vt,
                            int Tn, int Hn, size_t n) {
  size_t i = (size_t)blockIdx.x * 256 + threadIdx.x;
  if (i >= n) return;
  int e = (int)(i % 64);
  int h = (int)((i / 64) % Hn);
  int t = (int)((i / 512) % Tn);
  int b = (int)(i / ((size_t)512 * Tn));
  vt[(((size_t)b * Hn + h) * 64 + e) * Tn + t] = v[i];
}

// Attentive pooling: pooled[b, h*64+e] = sum_t softmax_t(logit[b,h,t]) * feat[b,t,h,e]
__global__ __launch_bounds__(256)
void pool_kernel(const float* __restrict__ logits /*[R,8]*/,
                 const float* __restrict__ feat   /*[R,512]*/,
                 const int* __restrict__ lens, float* __restrict__ pooled,
                 int Tn, int Hn) {
  const int b = blockIdx.x / Hn, h = blockIdx.x % Hn;
  const int tid = threadIdx.x;
  const int len = lens[b];
  __shared__ float red[256];
  __shared__ float ms[2];

  float mx = -3.0e38f;
  for (int t = tid; t < len; t += 256)
    mx = fmaxf(mx, logits[((size_t)b * Tn + t) * 8 + h]);
  red[tid] = mx;
  __syncthreads();
  for (int s = 128; s; s >>= 1) { if (tid < s) red[tid] = fmaxf(red[tid], red[tid + s]); __syncthreads(); }
  if (!tid) ms[0] = red[0];
  __syncthreads();
  const float m = ms[0];

  float sum = 0.0f;
  for (int t = tid; t < len; t += 256)
    sum += __expf(logits[((size_t)b * Tn + t) * 8 + h] - m);
  red[tid] = sum;
  __syncthreads();
  for (int s = 128; s; s >>= 1) { if (tid < s) red[tid] += red[tid + s]; __syncthreads(); }
  if (!tid) ms[1] = red[0];
  __syncthreads();
  const float stot = ms[1];

  const int e = tid & 63, slice = tid >> 6;
  float a = 0.0f;
  for (int t = slice; t < len; t += 4) {
    const float w = __expf(logits[((size_t)b * Tn + t) * 8 + h] - m);
    a += w * feat[((size_t)b * Tn + t) * 512 + h * 64 + e];
  }
  red[tid] = a;
  __syncthreads();
  if (slice == 0) {
    const float tot = red[e] + red[64 + e] + red[128 + e] + red[192 + e];
    pooled[(size_t)b * 512 + h * 64 + e] = tot / stot;
  }
}

// ---------------------------------------------------------------------------
// Host orchestration
// ---------------------------------------------------------------------------
extern "C" void kernel_launch(void* const* d_in, const int* in_sizes, int n_in,
                              void* d_out, int out_size, void* d_ws, size_t ws_size,
                              hipStream_t stream) {
  constexpr int Bn = 8, Tn = 1024, INn = 80, Hn = 8, Ln = 2;
  const int Rn = Bn * Tn;  // 8192 tokens

  const float* x  = (const float*)d_in[0];
  const int* lens = (const int*)d_in[1];
  auto F = [&](int i) -> const float* { return (const float*)d_in[i]; };

  // Bump allocator over d_ws (256B aligned).
  char* wsb = (char*)d_ws;
  size_t off = 0;
  auto alloc = [&](size_t bytes) -> void* {
    void* p = wsb + off;
    off += (bytes + 255) & ~(size_t)255;
    return p;
  };

  __bf16* xpad   = (__bf16*)alloc((size_t)Rn * 128 * 2);
  __bf16* ubf    = (__bf16*)alloc((size_t)Rn * 512 * 2);
  float*  t2f    = (float*) alloc((size_t)Rn * 512 * 4);
  __bf16* t2bf   = (__bf16*)alloc((size_t)Rn * 512 * 2);
  float*  hf     = (float*) alloc((size_t)Rn * 512 * 4);
  __bf16* hbf    = (__bf16*)alloc((size_t)Rn * 512 * 2);
  float*  h2f    = (float*) alloc((size_t)Rn * 512 * 4);
  __bf16* h2bf   = (__bf16*)alloc((size_t)Rn * 512 * 2);
  __bf16* qbf    = (__bf16*)alloc((size_t)Rn * 512 * 2);
  __bf16* kbf    = (__bf16*)alloc((size_t)Rn * 512 * 2);
  __bf16* vbf    = (__bf16*)alloc((size_t)Rn * 512 * 2);
  __bf16* vtbf   = (__bf16*)alloc((size_t)Rn * 512 * 2 + 256);  // pad for frag overreach
  __bf16* attbf  = (__bf16*)alloc((size_t)Rn * 512 * 2);
  float*  featf  = (float*) alloc((size_t)Rn * 512 * 4);
  float*  logitf = (float*) alloc((size_t)Rn * 8 * 4);
  float*  poolf  = (float*) alloc((size_t)Bn * 512 * 4);
  __bf16* poolbf = (__bf16*)alloc((size_t)Bn * 512 * 2);

  auto cvtw = [&](int idx, size_t n) -> const __bf16* {
    __bf16* dst = (__bf16*)alloc(n * 2);
    cvt_bf16<<<dim3((unsigned)((n + 255) / 256)), dim3(256), 0, stream>>>(F(idx), dst, n);
    return dst;
  };
  auto gemm = [&](const __bf16* A, const __bf16* W, const float* bias,
                  float* Cf, __bf16* Cb, int R_, int N_, int K_, int act) {
    if ((R_ % 256) == 0 && (N_ % 64) == 0) {
      dim3 g(R_ / 256, N_ / 64);
      gemm_full_wmma<<<g, dim3(256), 0, stream>>>(A, W, bias, Cf, Cb, R_, N_, K_, act);
    } else {
      dim3 g((R_ + 127) / 128, (N_ + 63) / 64);
      gemm_edge_wmma<<<g, dim3(256), 0, stream>>>(A, W, bias, Cf, Cb, R_, N_, K_, act);
    }
  };

  // Input: [R, 80] fp32 -> [R, 128] bf16 (K padded to multiple of 32).
  {
    size_t n = (size_t)Rn * 128;
    pad_cvt_bf16<<<dim3((unsigned)((n + 255) / 256)), dim3(256), 0, stream>>>(x, xpad, Rn, INn, 128);
  }

  const __bf16* cur = xpad;
  int Kin = 128;
  const size_t n512 = (size_t)Rn * 512;
  const unsigned gb512 = (unsigned)((n512 + 255) / 256);

  for (int l = 0; l < Ln; ++l) {
    const int pb = 2 + l * 30;  // per-layer leaves: see reference dict order
    const __bf16* th1;
    if (l == 0) {
      __bf16* w = (__bf16*)alloc((size_t)512 * 128 * 2);
      size_t n = (size_t)512 * 128;
      pad_cvt_bf16<<<dim3((unsigned)((n + 255) / 256)), dim3(256), 0, stream>>>(F(pb + 0), w, 512, INn, 128);
      th1 = w;
    } else {
      th1 = cvtw(pb + 0, 512 * 512);
    }
    const __bf16* th2 = cvtw(pb + 2, 512 * 512);
    const __bf16* q1  = cvtw(pb + 4, 512 * 512);
    const __bf16* q2  = cvtw(pb + 6, 512 * 512);
    const __bf16* qs  = cvtw(pb + 8, 512 * 512);   // [H,E,D] == [512,512]
    const __bf16* k1  = cvtw(pb + 10, 512 * 512);
    const __bf16* k2  = cvtw(pb + 12, 512 * 512);
    const __bf16* ks  = cvtw(pb + 14, 512 * 512);
    const __bf16* v1  = cvtw(pb + 16, 512 * 512);
    const __bf16* v2  = cvtw(pb + 18, 512 * 512);
    const __bf16* vs  = cvtw(pb + 20, 512 * 512);
    const __bf16* h2a = cvtw(pb + 22, 512 * 512);
    const __bf16* ff1 = cvtw(pb + 24, 512 * 512);
    const __bf16* ff2 = cvtw(pb + 26, 512 * 512);

    // h = to_hidden(x) + PE
    gemm(cur, th1, F(pb + 1), nullptr, ubf, Rn, 512, Kin, /*tanh*/1);
    gemm(ubf, th2, F(pb + 3), t2f, nullptr, Rn, 512, 512, 0);
    peadd_kernel<<<dim3(gb512), dim3(256), 0, stream>>>(t2f, hf, hbf, Tn, n512);

    // Q/K/V: MLP(tanh) -> split projection (block-diag over heads == dense 512x512)
    gemm(hbf, q1, F(pb + 5), nullptr, ubf, Rn, 512, 512, 1);
    gemm(ubf, q2, F(pb + 7), nullptr, t2bf, Rn, 512, 512, 0);
    gemm(t2bf, qs, F(pb + 9), nullptr, qbf, Rn, 512, 512, 0);

    gemm(hbf, k1, F(pb + 11), nullptr, ubf, Rn, 512, 512, 1);
    gemm(ubf, k2, F(pb + 13), nullptr, t2bf, Rn, 512, 512, 0);
    gemm(t2bf, ks, F(pb + 15), nullptr, kbf, Rn, 512, 512, 0);

    gemm(hbf, v1, F(pb + 17), nullptr, ubf, Rn, 512, 512, 1);
    gemm(ubf, v2, F(pb + 19), nullptr, t2bf, Rn, 512, 512, 0);
    gemm(t2bf, vs, F(pb + 21), nullptr, vbf, Rn, 512, 512, 0);

    transpose_v<<<dim3(gb512), dim3(256), 0, stream>>>(vbf, vtbf, Tn, Hn, n512);
    attn_flash_wmma<<<dim3(Tn / 64, Bn * Hn), dim3(128), 0, stream>>>(
        qbf, kbf, vtbf, lens, attbf, Bn, Hn, Tn);

    // h2 = LN(h + att @ h2a^T + b);  out = LN(h2 + FF(h2))
    gemm(attbf, h2a, F(pb + 23), t2f, nullptr, Rn, 512, 512, 0);
    ln_kernel<<<dim3(Rn), dim3(256), 0, stream>>>(hf, t2f, F(pb + 28), F(pb + 29), h2f, h2bf);
    gemm(h2bf, ff1, F(pb + 25), nullptr, ubf, Rn, 512, 512, /*relu*/2);
    gemm(ubf, ff2, F(pb + 27), t2f, nullptr, Rn, 512, 512, 0);
    ln_kernel<<<dim3(Rn), dim3(256), 0, stream>>>(h2f, t2f, F(pb + 28), F(pb + 29), hf, hbf);

    cur = hbf;
    Kin = 512;
  }

  // Attentive pooling head.
  const int hb = 2 + Ln * 30;
  const __bf16* fm1   = cvtw(hb + 0, 512 * 512);
  const __bf16* fm2   = cvtw(hb + 2, 512 * 512);
  const __bf16* fsw   = cvtw(hb + 4, 512 * 512);  // ff_split w [H,E,D]
  const __bf16* fw1   = cvtw(hb + 6, 512 * 512);
  const __bf16* fw2   = cvtw(hb + 8, 512 * 512);
  const __bf16* fww   = cvtw(hb + 10, 8 * 512);   // fw_split w [H,1,D]
  const __bf16* lastw = cvtw(hb + 12, 512 * 512);

  gemm(cur, fm1, F(hb + 1), nullptr, ubf, Rn, 512, 512, 1);
  gemm(ubf, fm2, F(hb + 3), nullptr, t2bf, Rn, 512, 512, 0);
  gemm(t2bf, fsw, F(hb + 5), featf, nullptr, Rn, 512, 512, 0);  // feat [R, H*E]

  gemm(cur, fw1, F(hb + 7), nullptr, ubf, Rn, 512, 512, 1);
  gemm(ubf, fw2, F(hb + 9), nullptr, t2bf, Rn, 512, 512, 0);
  gemm(t2bf, fww, F(hb + 11), logitf, nullptr, Rn, 8, 512, 0);  // logit [R, H]

  pool_kernel<<<dim3(Bn * Hn), dim3(256), 0, stream>>>(logitf, featf, lens, poolf, Tn, Hn);
  cvt_bf16<<<dim3((Bn * 512 + 255) / 256), dim3(256), 0, stream>>>(poolf, poolbf, (size_t)Bn * 512);

  // out [B, 512] = pooled @ last_w^T + b  (row-guarded WMMA GEMM writes d_out)
  gemm(poolbf, lastw, F(hb + 13), (float*)d_out, nullptr, Bn, 512, 512, 0);
}
